// SimplifiedGPT_86689619903517
// MI455X (gfx1250) — matrix operand order repaired and checked by reading, not compiled, not run on hardware
//
#include <hip/hip_runtime.h>

typedef __attribute__((ext_vector_type(16))) __bf16 v16bf;
typedef __attribute__((ext_vector_type(8)))  __bf16 v8bf;
typedef __attribute__((ext_vector_type(4)))  __bf16 v4bf;
typedef __attribute__((ext_vector_type(8)))  float  v8f;

#define VOCAB 8192
#define C_DIM 1024
#define L_NUM 4
#define B_NUM 8
#define T_NUM 1024
#define M_TOK (B_NUM * T_NUM)   // 8192 token rows

// ---------------------------------------------------------------------------
// Embedding gather: x[m, :] = (bf16) wte[idx[m], :]   (vectorized by 4)
// ---------------------------------------------------------------------------
__global__ __launch_bounds__(256)
void embed_kernel(const int* __restrict__ idx, const float* __restrict__ wte,
                  __bf16* __restrict__ x) {
    size_t i = (size_t)blockIdx.x * 256 + threadIdx.x;   // over M_TOK * 256
    size_t m  = i >> 8;            // C_DIM/4 == 256 chunks per row
    int    c4 = (int)(i & 255);
    const float4 w = ((const float4*)(wte + (size_t)idx[m] * C_DIM))[c4];
    v4bf o;
    o[0] = (__bf16)w.x; o[1] = (__bf16)w.y; o[2] = (__bf16)w.z; o[3] = (__bf16)w.w;
    *(v4bf*)(x + m * C_DIM + (size_t)c4 * 4) = o;
}

// ---------------------------------------------------------------------------
// f32 -> bf16 weight conversion (vectorized by 4)
// ---------------------------------------------------------------------------
__global__ __launch_bounds__(256)
void cvt_bf16_kernel(const float* __restrict__ in, __bf16* __restrict__ out, int n4) {
    int i = blockIdx.x * 256 + threadIdx.x;
    if (i < n4) {
        float4 w = ((const float4*)in)[i];
        v4bf o;
        o[0] = (__bf16)w.x; o[1] = (__bf16)w.y; o[2] = (__bf16)w.z; o[3] = (__bf16)w.w;
        ((v4bf*)out)[i] = o;
    }
}

// ---------------------------------------------------------------------------
// Value projection collapsed to t==1 row:  v1[b,d] = sum_c x[b,1,c] * wv[d,c]
// ---------------------------------------------------------------------------
__global__ __launch_bounds__(256)
void v1_kernel(const __bf16* __restrict__ x, const float* __restrict__ wvl,
               float* __restrict__ v1) {
    int o = blockIdx.x * 256 + threadIdx.x;     // 0 .. B*C-1 (8192)
    int b = o >> 10, d = o & 1023;
    const __bf16* xr = x + ((size_t)(b * T_NUM + 1)) * C_DIM;
    const float*  wr = wvl + (size_t)d * C_DIM;
    float s = 0.f;
    #pragma unroll 4
    for (int c = 0; c < C_DIM; ++c) s += (float)xr[c] * wr[c];
    v1[o] = s;
}

// ---------------------------------------------------------------------------
// WMMA fragment loaders (bf16, K-contiguous row-major sources, stride C_DIM)
// A 16x32: lane m&15 -> row; lanes<16: K k0..k0+15, lanes>=16: K k0+16..k0+31
// B 32x16: lane n&15 -> weight row (output col); K-contiguous 16 halves
// ---------------------------------------------------------------------------
union FragU { v16bf v; v8bf h[2]; };

__device__ __forceinline__ v16bf load_a_frag(const __bf16* base, int k0, int lane) {
    const int r  = lane & 15;
    const int lo = (lane & 16) ? 16 : 0;
    const __bf16* p = base + (size_t)r * C_DIM + k0 + lo;
    FragU f;
    f.h[0] = *(const v8bf*)(p);        // K = k0+lo .. +7
    f.h[1] = *(const v8bf*)(p + 8);    // K = k0+lo+8 .. +15
    return f.v;
}

__device__ __forceinline__ v16bf load_b_frag(const __bf16* base, int k0, int lane) {
    const int n  = lane & 15;
    const int lo = (lane & 16) ? 16 : 0;
    return *(const v16bf*)(base + (size_t)n * C_DIM + k0 + lo);   // 32B contiguous
}

// ---------------------------------------------------------------------------
// GEMM: out[m,n] = sum_k A[m,k] * W[n,k],  K = C_DIM = 1024.
// Block = 4 waves (128 thr) in a 2x2 wave grid; block tile 64x128.
// Wave tile 32x64 = 2 M-subtiles x 4 N-subtiles -> 8 WMMAs per 12 b128 loads.
// EPILOGUE 0: f32 store (logits).  EPILOGUE 1: +v1[b] broadcast, bf16 store.
// ---------------------------------------------------------------------------
template <int EPILOGUE>
__global__ __launch_bounds__(128)
void gemm_wmma(const __bf16* __restrict__ A, const __bf16* __restrict__ W, int N,
               const float* __restrict__ v1, __bf16* __restrict__ outb,
               float* __restrict__ outf) {
    const int lane = threadIdx.x & 31;
    const int wave = threadIdx.x >> 5;
    const int m0 = blockIdx.x * 64  + (wave >> 1) * 32;   // 2 waves along M
    const int n0 = blockIdx.y * 128 + (wave & 1) * 64;    // 2 waves along N

    const __bf16* ap[2];
    const __bf16* bp[4];
    #pragma unroll
    for (int mt = 0; mt < 2; ++mt) ap[mt] = A + (size_t)(m0 + mt * 16) * C_DIM;
    #pragma unroll
    for (int nt = 0; nt < 4; ++nt) bp[nt] = W + (size_t)(n0 + nt * 16) * C_DIM;

    v8f acc[2][4];
    #pragma unroll
    for (int mt = 0; mt < 2; ++mt)
        #pragma unroll
        for (int nt = 0; nt < 4; ++nt) acc[mt][nt] = (v8f){};

    #pragma unroll 2
    for (int k0 = 0; k0 < C_DIM; k0 += 32) {
        v16bf a[2], b[4];
        #pragma unroll
        for (int mt = 0; mt < 2; ++mt) a[mt] = load_a_frag(ap[mt], k0, lane);
        #pragma unroll
        for (int nt = 0; nt < 4; ++nt) b[nt] = load_b_frag(bp[nt], k0, lane);
        #pragma unroll
        for (int mt = 0; mt < 2; ++mt)
            #pragma unroll
            for (int nt = 0; nt < 4; ++nt)
                acc[mt][nt] = __builtin_amdgcn_wmma_f32_16x16x32_bf16(
                    false, a[mt], false, b[nt], (short)0, acc[mt][nt], false, false);
    }

    // D layout: VGPR r -> M = r + 8*(lane>=16), N = lane&15
    const int col   = lane & 15;
    const int rbase = (lane & 16) ? 8 : 0;
    #pragma unroll
    for (int mt = 0; mt < 2; ++mt) {
        #pragma unroll
        for (int r = 0; r < 8; ++r) {
            const int row = m0 + mt * 16 + rbase + r;
            const int bi  = row >> 10;          // token row -> batch (T = 1024)
            #pragma unroll
            for (int nt = 0; nt < 4; ++nt) {
                const int gc = n0 + nt * 16 + col;
                if (EPILOGUE == 1) {
                    outb[(size_t)row * N + gc] =
                        (__bf16)(acc[mt][nt][r] + v1[bi * C_DIM + gc]);
                } else {
                    outf[(size_t)row * N + gc] = acc[mt][nt][r];
                }
            }
        }
    }
}

// ---------------------------------------------------------------------------
// Cross entropy: one block per token row; masked (target != 0) NLL into acc.
// ---------------------------------------------------------------------------
__global__ void init_acc_kernel(float* acc) {
    if (threadIdx.x < 2) acc[threadIdx.x] = 0.f;
}

__global__ __launch_bounds__(256)
void rowloss_kernel(const float* __restrict__ logits, const int* __restrict__ targets,
                    float* __restrict__ acc) {
    const int row = blockIdx.x;
    const float* p = logits + (size_t)row * VOCAB;
    __shared__ float red[256];
    const int tid = threadIdx.x;

    float mx = -1e30f;
    for (int c = tid; c < VOCAB; c += 256) mx = fmaxf(mx, p[c]);
    red[tid] = mx; __syncthreads();
    for (int s = 128; s > 0; s >>= 1) {
        if (tid < s) red[tid] = fmaxf(red[tid], red[tid + s]);
        __syncthreads();
    }
    const float m = red[0]; __syncthreads();

    float sum = 0.f;
    for (int c = tid; c < VOCAB; c += 256) sum += __expf(p[c] - m);
    red[tid] = sum; __syncthreads();
    for (int s = 128; s > 0; s >>= 1) {
        if (tid < s) red[tid] += red[tid + s];
        __syncthreads();
    }

    if (tid == 0) {
        const int t = targets[row];
        if (t != 0) {
            const float nll = -(p[t] - m - __logf(red[0]));
            atomicAdd(acc + 0, nll);
            atomicAdd(acc + 1, 1.0f);
        }
    }
}

__global__ void finalize_kernel(const float* __restrict__ acc, float* __restrict__ loss) {
    if (threadIdx.x == 0) loss[0] = acc[0] / fmaxf(acc[1], 1.0f);
}

// ---------------------------------------------------------------------------
// Launcher
// ---------------------------------------------------------------------------
extern "C" void kernel_launch(void* const* d_in, const int* in_sizes, int n_in,
                              void* d_out, int out_size, void* d_ws, size_t ws_size,
                              hipStream_t stream) {
    (void)in_sizes; (void)n_in; (void)out_size; (void)ws_size;
    const int*   idx  = (const int*)d_in[0];
    const int*   tgt  = (const int*)d_in[1];
    const float* wte  = (const float*)d_in[2];
    const float* wv   = (const float*)d_in[3];
    const float* wm   = (const float*)d_in[4];
    const float* lm   = (const float*)d_in[5];
    float* out = (float*)d_out;                 // [M_TOK*VOCAB logits][1 loss]

    // Workspace carve (all offsets 256B-aligned)
    char* p = (char*)d_ws;
    __bf16* xA  = (__bf16*)(p);                                  // 16 MB
    __bf16* xB  = (__bf16*)(p + (size_t)16 * 1024 * 1024);       // 16 MB
    __bf16* wmB = (__bf16*)(p + (size_t)32 * 1024 * 1024);       //  8 MB
    __bf16* lmB = (__bf16*)(p + (size_t)40 * 1024 * 1024);       // 16 MB
    float*  v1  = (float*) (p + (size_t)56 * 1024 * 1024);       // 32 KB
    float*  acc = (float*) (p + (size_t)56 * 1024 * 1024 + 65536);

    // 1) embedding gather -> bf16 activations
    embed_kernel<<<M_TOK, 256, 0, stream>>>(idx, wte, xA);

    // 2) weight down-conversion f32 -> bf16 (stays hot in 192MB L2)
    cvt_bf16_kernel<<<(L_NUM * C_DIM * C_DIM / 4 + 255) / 256, 256, 0, stream>>>(
        wm, wmB, L_NUM * C_DIM * C_DIM / 4);
    cvt_bf16_kernel<<<(VOCAB * C_DIM / 4 + 255) / 256, 256, 0, stream>>>(
        lm, lmB, VOCAB * C_DIM / 4);

    // 3) transformer layers: tiny v-projection (t==1 only) + fused FFN GEMM
    __bf16* xc = xA;
    __bf16* xn = xB;
    for (int l = 0; l < L_NUM; ++l) {
        v1_kernel<<<B_NUM * C_DIM / 256, 256, 0, stream>>>(
            xc, wv + (size_t)l * C_DIM * C_DIM, v1);
        gemm_wmma<1><<<dim3(M_TOK / 64, C_DIM / 128), 128, 0, stream>>>(
            xc, wmB + (size_t)l * C_DIM * C_DIM, C_DIM, v1, xn, nullptr);
        __bf16* t = xc; xc = xn; xn = t;
    }

    // 4) LM head -> f32 logits straight to d_out
    gemm_wmma<0><<<dim3(M_TOK / 64, VOCAB / 128), 128, 0, stream>>>(
        xc, lmB, VOCAB, nullptr, nullptr, out);

    // 5) masked mean cross-entropy
    init_acc_kernel<<<1, 32, 0, stream>>>(acc);
    rowloss_kernel<<<M_TOK, 256, 0, stream>>>(out, tgt, acc);
    finalize_kernel<<<1, 32, 0, stream>>>(acc, out + (size_t)M_TOK * VOCAB);
}
